// FPSPool_2508260901456
// MI455X (gfx1250) — compile-verified, matrix operand order checked by reading, not run on hardware
//
#include <hip/hip_runtime.h>
#include <math.h>
#include <stdint.h>

// Problem constants (match the JAX reference)
#define B_    4
#define N_    32768
#define CIN_  128
#define COUT_ 256
#define S_    2048
#define M_    (B_ * N_)        // 131072 rows for BN statistics
#define EPS_  1e-5f

typedef __attribute__((ext_vector_type(16))) _Float16 v16h;
typedef __attribute__((ext_vector_type(8)))  float    v8f;

// ---------------------------------------------------------------------------
// Fragment loader: convert a contiguous-K fp32 row segment into the wave32
// 16-bit WMMA operand striping (ISA 7.12.2):
//   lane l holds row/col (l & 15); kb = (l>>4)*8 selects the K half-chunk.
//   elements 0..7  <-> K = kb + e
//   elements 8..15 <-> K = kb + 16 + e
// rowp must point at (row base + k0).  Works for global or LDS pointers.
// ---------------------------------------------------------------------------
__device__ __forceinline__ v16h load_frag_f32_to_f16(const float* rowp, int kb) {
  v16h f;
#pragma unroll
  for (int e = 0; e < 8; ++e) {
    f[e]     = (_Float16)rowp[kb + e];
    f[e + 8] = (_Float16)rowp[kb + 16 + e];
  }
  return f;
}

// ---------------------------------------------------------------------------
// Zero the 512-float stats accumulators (must run every call: harness does
// not re-clear workspace between replays).
// ---------------------------------------------------------------------------
__global__ __launch_bounds__(512)
void init_stats_kernel(float* __restrict__ acc) {
  acc[threadIdx.x] = 0.0f;
}

// ---------------------------------------------------------------------------
// Stats GEMM: compute per-column sum(x) and sum(x^2) of x = feat @ W.T
// without materializing x.  Grid: 2048 blocks x 256 threads (8 waves).
//   blockIdx.x & 15  -> output column tile (tile_n)
//   blockIdx.x >> 4  -> M-tile group; each wave sweeps 8 M-tiles.
// The block's 16 W rows (16x128 f32 = 8 KB) are staged into LDS with the
// CDNA5 async copy engine (GLOBAL_LOAD_ASYNC_TO_LDS_B128, ASYNCcnt), then
// each wave computes 16x16 f32 tiles via v_wmma_f32_16x16x32_f16.
// ---------------------------------------------------------------------------
__global__ __launch_bounds__(256)
void stats_gemm_kernel(const float* __restrict__ feat,
                       const float* __restrict__ W,
                       float* __restrict__ g_sum,
                       float* __restrict__ g_sq) {
  __shared__ float wtile[16 * CIN_];           // 8 KB: this block's W rows

  const int wave   = threadIdx.x >> 5;
  const int l      = threadIdx.x & 31;
  const int kb     = (l >> 4) * 8;
  const int tile_n = blockIdx.x & 15;
  const int mgrp   = blockIdx.x >> 4;          // 0..127

  // --- Async-stage W[tile_n*16 .. tile_n*16+16) into LDS ------------------
  // GVS addressing: per-lane 16B copy, LDS byte offset in vdst operand
  // (low 32 bits of the generic shared address == LDS offset, ISA 10.2),
  // 32-bit per-lane offset VGPR + 64-bit SGPR base.
  {
    const float* wsrc = W + (size_t)tile_n * 16 * CIN_;
    const unsigned lbase = (unsigned)(uintptr_t)(&wtile[0]);
    const unsigned off0  = (unsigned)threadIdx.x * 16u;
#pragma unroll
    for (unsigned p = 0; p < 2; ++p) {
      unsigned ldsdst = lbase + off0 + p * 4096u;
      unsigned voff   = off0 + p * 4096u;
      asm volatile("global_load_async_to_lds_b128 %0, %1, %2"
                   :: "v"(ldsdst), "v"(voff), "s"(wsrc)
                   : "memory");
    }
    asm volatile("s_wait_asynccnt 0" ::: "memory");
  }
  __syncthreads();

  // B fragments come from LDS; they depend only on (lane, k) -> hoist.
  const int   nloc = (l & 15);
  const int   ncol = tile_n * 16 + nloc;
  const float* brow = &wtile[nloc * CIN_];
  v16h bf[4];
#pragma unroll
  for (int kk = 0; kk < 4; ++kk) bf[kk] = load_frag_f32_to_f16(brow + kk * 32, kb);

  float psum = 0.0f, psq = 0.0f;

  for (int it = 0; it < 8; ++it) {
    const int m_tile = (mgrp * 8 + wave) + 1024 * it;   // 0..8191
    const int mrow   = m_tile * 16 + nloc;
    const float* arow = feat + (size_t)mrow * CIN_;

    // Prefetch next iteration's A row into the cache hierarchy.
    if (it < 7) __builtin_prefetch(feat + (size_t)(mrow + 1024 * 16) * CIN_, 0, 1);

    v8f acc = {};
#pragma unroll
    for (int kk = 0; kk < 4; ++kk) {
      v16h af = load_frag_f32_to_f16(arow + kk * 32, kb);
      acc = __builtin_amdgcn_wmma_f32_16x16x32_f16(
          /*neg_a=*/false, af, /*neg_b=*/false, bf[kk],
          /*c_mod=*/(short)0, acc, /*reuse_a=*/false, /*reuse_b=*/false);
    }
    // C/D layout: VGPR i holds rows, lane gives fixed column ncol.
#pragma unroll
    for (int i = 0; i < 8; ++i) {
      float v = acc[i];
      psum += v;
      psq  += v * v;
    }
  }

  // Lane l and lane l+16 hold the same column -> combine the halves.
  psum += __shfl_xor(psum, 16, 32);
  psq  += __shfl_xor(psq, 16, 32);

  __shared__ float ssum[16];
  __shared__ float ssq[16];
  if (threadIdx.x < 16) { ssum[threadIdx.x] = 0.0f; ssq[threadIdx.x] = 0.0f; }
  __syncthreads();
  if (l < 16) {               // one contribution per wave per column
    atomicAdd(&ssum[l], psum);
    atomicAdd(&ssq[l],  psq);
  }
  __syncthreads();
  if (threadIdx.x < 16) {
    atomicAdd(&g_sum[tile_n * 16 + threadIdx.x], ssum[threadIdx.x]);
    atomicAdd(&g_sq [tile_n * 16 + threadIdx.x], ssq [threadIdx.x]);
  }
}

// ---------------------------------------------------------------------------
// Farthest point sampling: one 1024-thread workgroup per batch (sequential
// 2048-step scan; coords are L2-resident).  Each thread owns 32 points'
// running min-distance in VGPRs; argmax via wave32 shuffle tree + LDS.
// Mirrors the reference scan: record current farthest, then update.
// Also writes the sampled coords and the (trivial) new offsets.
// ---------------------------------------------------------------------------
__global__ __launch_bounds__(1024)
void fps_kernel(const float* __restrict__ coord,
                int*   __restrict__ idx_out,     // [B*S]
                float* __restrict__ out_coords,  // [B*S*3]
                float* __restrict__ out_off) {   // [B]
  const int b = blockIdx.x;
  const int t = threadIdx.x;
  const float* cb = coord + (size_t)b * N_ * 3;

  if (b == 0 && t < B_) out_off[t] = (float)((t + 1) * S_);

  float dist[32];
#pragma unroll
  for (int i = 0; i < 32; ++i) dist[i] = 1e10f;

  __shared__ int   s_far;
  __shared__ float s_wval[32];
  __shared__ int   s_widx[32];
  if (t == 0) s_far = 0;                 // deterministic start index 0
  __syncthreads();

  for (int it = 0; it < S_; ++it) {
    const int far = s_far;

    if (t == 0) {
      idx_out[b * S_ + it] = far;
      out_coords[(size_t)(b * S_ + it) * 3 + 0] = cb[(size_t)far * 3 + 0];
      out_coords[(size_t)(b * S_ + it) * 3 + 1] = cb[(size_t)far * 3 + 1];
      out_coords[(size_t)(b * S_ + it) * 3 + 2] = cb[(size_t)far * 3 + 2];
    }

    const float fx = cb[(size_t)far * 3 + 0];
    const float fy = cb[(size_t)far * 3 + 1];
    const float fz = cb[(size_t)far * 3 + 2];

    float bestv = -1.0f;
    int   besti = 0;
#pragma unroll
    for (int i = 0; i < 32; ++i) {
      const int p = i * 1024 + t;
      const float dx = cb[(size_t)p * 3 + 0] - fx;
      const float dy = cb[(size_t)p * 3 + 1] - fy;
      const float dz = cb[(size_t)p * 3 + 2] - fz;
      const float d  = dx * dx + dy * dy + dz * dz;
      const float nd = fminf(dist[i], d);
      dist[i] = nd;
      if (nd > bestv) { bestv = nd; besti = p; }   // ascending p -> first max kept
    }

    // wave32 argmax reduce (lowest index wins ties, like jnp.argmax)
#pragma unroll
    for (int off = 16; off > 0; off >>= 1) {
      const float ov = __shfl_xor(bestv, off, 32);
      const int   oi = __shfl_xor(besti, off, 32);
      if (ov > bestv || (ov == bestv && oi < besti)) { bestv = ov; besti = oi; }
    }
    const int w = t >> 5;
    if ((t & 31) == 0) { s_wval[w] = bestv; s_widx[w] = besti; }
    __syncthreads();
    if (w == 0) {
      bestv = s_wval[t & 31];
      besti = s_widx[t & 31];
#pragma unroll
      for (int off = 16; off > 0; off >>= 1) {
        const float ov = __shfl_xor(bestv, off, 32);
        const int   oi = __shfl_xor(besti, off, 32);
        if (ov > bestv || (ov == bestv && oi < besti)) { bestv = ov; besti = oi; }
      }
      if (t == 0) s_far = besti;
    }
    __syncthreads();
  }
}

// ---------------------------------------------------------------------------
// Gather + FC + BN + ReLU: recompute x only for the 8192 sampled rows, fused
// with BatchNorm (stats from g_sum/g_sq) and ReLU.  8192 wave-tiles of 16x16:
// 1024 blocks x 8 waves.
// ---------------------------------------------------------------------------
__global__ __launch_bounds__(256)
void gather_fc_bn_kernel(const float* __restrict__ feat,
                         const float* __restrict__ W,
                         const float* __restrict__ gamma,
                         const float* __restrict__ beta,
                         const int*   __restrict__ idxbuf,   // [B*S] local idx
                         const float* __restrict__ g_sum,
                         const float* __restrict__ g_sq,
                         float* __restrict__ out_feat) {     // [B*S, COUT]
  const int wave = threadIdx.x >> 5;
  const int l    = threadIdx.x & 31;
  const int kb   = (l >> 4) * 8;

  const int tile_id = blockIdx.x * 8 + wave;   // 0..8191
  const int tile_m  = tile_id >> 4;            // 0..511
  const int tile_n  = tile_id & 15;            // 0..15

  const int mrow = tile_m * 16 + (l & 15);     // sampled-row index, 0..8191
  const int bb   = mrow / S_;
  const int loc  = idxbuf[mrow];               // local point index within batch
  const float* arow = feat + ((size_t)bb * N_ + loc) * CIN_;

  const int ncol = tile_n * 16 + (l & 15);
  const float* brow = W + (size_t)ncol * CIN_;

  v8f acc = {};
#pragma unroll
  for (int kk = 0; kk < 4; ++kk) {
    v16h af = load_frag_f32_to_f16(arow + kk * 32, kb);
    v16h bf = load_frag_f32_to_f16(brow + kk * 32, kb);
    acc = __builtin_amdgcn_wmma_f32_16x16x32_f16(
        false, af, false, bf, (short)0, acc, false, false);
  }

  // BatchNorm (biased variance, training-mode batch stats) + affine + ReLU
  const float inv_m = 1.0f / (float)M_;
  const float mean  = g_sum[ncol] * inv_m;
  const float var   = g_sq[ncol] * inv_m - mean * mean;
  const float scale = gamma[ncol] * rsqrtf(var + EPS_);
  const float shift = beta[ncol] - mean * scale;

  const int hi = l >> 4;
#pragma unroll
  for (int i = 0; i < 8; ++i) {
    const int m = tile_m * 16 + i + 8 * hi;
    const float v = fmaxf(acc[i] * scale + shift, 0.0f);
    out_feat[(size_t)m * COUT_ + ncol] = v;
  }
}

// ---------------------------------------------------------------------------
// Host-side launcher.  d_in order: coord, feat, offset, W, gamma, beta.
// d_out: [new_coords (B*S*3) | new_feats (B*S*COUT) | new_offsets (B)].
// ---------------------------------------------------------------------------
extern "C" void kernel_launch(void* const* d_in, const int* in_sizes, int n_in,
                              void* d_out, int out_size, void* d_ws, size_t ws_size,
                              hipStream_t stream) {
  const float* coord = (const float*)d_in[0];
  const float* feat  = (const float*)d_in[1];
  // d_in[2] = offset (int32) — values implied by B_/N_, unused
  const float* W     = (const float*)d_in[3];
  const float* gamma = (const float*)d_in[4];
  const float* beta  = (const float*)d_in[5];

  // Workspace: 256 col-sums, 256 col-sumsq, B*S sampled indices.
  float* g_sum = (float*)d_ws;
  float* g_sq  = g_sum + 256;
  int*   idxb  = (int*)(g_sq + 256);

  float* out_coords = (float*)d_out;                         // B*S*3
  float* out_feat   = out_coords + (size_t)B_ * S_ * 3;      // B*S*COUT
  float* out_off    = out_feat + (size_t)B_ * S_ * COUT_;    // B

  hipLaunchKernelGGL(init_stats_kernel, dim3(1), dim3(512), 0, stream, g_sum);
  hipLaunchKernelGGL(stats_gemm_kernel, dim3(2048), dim3(256), 0, stream,
                     feat, W, g_sum, g_sq);
  hipLaunchKernelGGL(fps_kernel, dim3(B_), dim3(1024), 0, stream,
                     coord, idxb, out_coords, out_off);
  hipLaunchKernelGGL(gather_fc_bn_kernel, dim3(1024), dim3(256), 0, stream,
                     feat, W, gamma, beta, idxb, g_sum, g_sq, out_feat);
}
